// RecurrentRGCN_53661321396270
// MI455X (gfx1250) — compile-verified, matrix operand order checked by reading, not run on hardware
//
#include <hip/hip_runtime.h>
#include <hip/hip_bf16.h>

typedef __attribute__((ext_vector_type(16))) __bf16 v16bf;
typedef __attribute__((ext_vector_type(8)))  float  v8f;

#define E_N   50000
#define D_N   200
#define T_N   3
#define NE_N  400000
#define NR2_N 460
#define RRELU_SLOPE 0.2291666666666667f

// ---------------------------------------------------------------------------
// Row-wise L2 normalize: one wave32 per row.
// ---------------------------------------------------------------------------
__global__ __launch_bounds__(256) void l2norm_rows(const float* __restrict__ src,
                                                   float* __restrict__ dst, int rows) {
    int warp = (blockIdx.x * blockDim.x + threadIdx.x) >> 5;
    int lane = threadIdx.x & 31;
    if (warp >= rows) return;
    const float* x = src + (size_t)warp * D_N;
    float s = 0.f;
    for (int j = lane; j < D_N; j += 32) { float v = x[j]; s += v * v; }
    #pragma unroll
    for (int o = 16; o > 0; o >>= 1) s += __shfl_xor(s, o, 32);
    float inv = rsqrtf(s + 1e-12f);
    float* y = dst + (size_t)warp * D_N;
    for (int j = lane; j < D_N; j += 32) y[j] = x[j] * inv;
}

// ---------------------------------------------------------------------------
// Edge scatter: agg[dst] += h[src] + rel[etype]; optionally deg[dst] += 1.
// One wave32 per edge; fp32 HW atomics (global_atomic_add_f32, no return).
// All 40MB node buffers are L2-resident (192MB), so the random gather/scatter
// runs at L2 rather than HBM latency.
// ---------------------------------------------------------------------------
__global__ __launch_bounds__(256) void edge_agg(const float* __restrict__ h,
                                                const float* __restrict__ rel,
                                                const int* __restrict__ src,
                                                const int* __restrict__ dst,
                                                const int* __restrict__ et,
                                                float* __restrict__ agg,
                                                float* __restrict__ deg,
                                                int ne, int addDeg) {
    int warp = (blockIdx.x * blockDim.x + threadIdx.x) >> 5;
    int lane = threadIdx.x & 31;
    if (warp >= ne) return;
    int s = src[warp], d = dst[warp], t = et[warp];
    const float* hs = h + (size_t)s * D_N;
    const float* rt = rel + (size_t)t * D_N;
    float* ag = agg + (size_t)d * D_N;
    for (int j = lane; j < D_N; j += 32)
        unsafeAtomicAdd(&ag[j], hs[j] + rt[j]);
    if (addDeg && lane == 0) unsafeAtomicAdd(&deg[d], 1.0f);
}

// ---------------------------------------------------------------------------
// bf16 WMMA GEMM: C[MxN] = epilogue(A[MxK] @ B[KxN]).
//   transB=1 : B_eff[k][n] = B[n*ldb + k]   (for W^T GRU slices)
//   mode=0   : plain (+bias[n] if bias != null)
//   mode=1   : RGCN epilogue  v *= (deg[m]>0 ? 1/deg[m] : 0); rrelu
// 4 waves/block -> 64x64 output tile. Each wave owns one 16-col N subtile and
// 4 M subtiles (4 f32 accumulators); the B fragment is packed once per K-step
// and reused across all 4 WMMAs (16 WMMAs per block per K-step). A/B staged
// f32 in LDS, packed to bf16 per the CDNA5 16-bit fragment layouts.
// ---------------------------------------------------------------------------
__global__ __launch_bounds__(128) void gemm_bf16_wmma(
    const float* __restrict__ A, const float* __restrict__ B, float* __restrict__ C,
    int M, int N, int K, int ldb, int transB, int mode,
    const float* __restrict__ deg, const float* __restrict__ bias, float slope) {

    __shared__ float lA[64 * 32];   // 8 KB
    __shared__ float lB[32 * 64];   // 8 KB

    const int tid  = threadIdx.x;
    const int lane = tid & 31;
    const int wv   = tid >> 5;          // wave 0..3 -> N sub-tile
    const int m0   = blockIdx.y * 64;
    const int n0   = blockIdx.x * 64;

    v8f acc[4];
    #pragma unroll
    for (int ms = 0; ms < 4; ++ms)
        acc[ms] = (v8f){0.f, 0.f, 0.f, 0.f, 0.f, 0.f, 0.f, 0.f};

    for (int k0 = 0; k0 < K; k0 += 32) {
        // cooperative A tile load (64x32), zero-fill M/K tails
        for (int i = tid; i < 64 * 32; i += 128) {
            int r = i >> 5, c = i & 31;
            int m = m0 + r, k = k0 + c;
            lA[i] = (m < M && k < K) ? A[(size_t)m * K + k] : 0.f;
        }
        // cooperative B tile load (32x64), zero-fill K/N tails
        for (int i = tid; i < 32 * 64; i += 128) {
            int kk = i >> 6, c = i & 63;
            int k = k0 + kk, n = n0 + c;
            float v = 0.f;
            if (k < K && n < N)
                v = transB ? B[(size_t)n * ldb + k] : B[(size_t)k * ldb + n];
            lB[i] = v;
        }
        __syncthreads();

        // prefetch next A K-slice into cache while WMMAs run
        if (k0 + 32 < K && tid < 64 && (m0 + tid) < M)
            __builtin_prefetch(&A[(size_t)(m0 + tid) * K + k0 + 32], 0, 1);

        // pack B fragment once (CDNA5 16-bit 32x16 layout, mirrored on N)
        const int mloc = lane & 15;
        const int half = lane >> 4;
        v16bf b;
        #pragma unroll
        for (int e = 0; e < 16; ++e) {
            int v = e >> 1, p = e & 1;
            int kk = ((v >> 2) << 4) + (half << 3) + ((v & 3) << 1) + p;
            b[e] = (__bf16)lB[kk * 64 + (wv << 4) + mloc];
        }
        // 4 M-subtiles reuse the same B fragment
        #pragma unroll
        for (int ms = 0; ms < 4; ++ms) {
            v16bf a;
            #pragma unroll
            for (int e = 0; e < 16; ++e) {
                int v = e >> 1, p = e & 1;
                int kk = ((v >> 2) << 4) + (half << 3) + ((v & 3) << 1) + p;
                a[e] = (__bf16)lA[((ms << 4) + mloc) * 32 + kk];
            }
            acc[ms] = __builtin_amdgcn_wmma_f32_16x16x32_bf16(
                          false, a, false, b, (short)0, acc[ms], false, false);
        }
        __syncthreads();
    }

    // epilogue + store (C/D layout: VGPR r -> M = r + 8*(lane>=16), N = lane&15)
    const int n = n0 + (wv << 4) + (lane & 15);
    if (n < N) {
        const int mh = (lane >> 4) << 3;
        #pragma unroll
        for (int ms = 0; ms < 4; ++ms) {
            #pragma unroll
            for (int r = 0; r < 8; ++r) {
                int m = m0 + (ms << 4) + r + mh;
                if (m >= M) continue;
                float v = acc[ms][r];
                if (mode == 1) {
                    float dg = deg[m];
                    float nm = dg > 0.f ? 1.0f / dg : 0.f;
                    v *= nm;
                    v = v >= 0.f ? v : v * slope;
                } else if (bias) {
                    v += bias[n];
                }
                C[(size_t)m * N + n] = v;
            }
        }
    }
}

// ---------------------------------------------------------------------------
// Elementwise fusion kernels
// ---------------------------------------------------------------------------
__device__ __forceinline__ float sigm(float x) { return 1.0f / (1.0f + __expf(-x)); }

__global__ void ew_sigadd(const float* a, const float* b, float* o, int n) {
    int i = blockIdx.x * blockDim.x + threadIdx.x;
    if (i < n) o[i] = sigm(a[i] + b[i]);
}
// n = tanh(inn + r*hn)
__global__ void ew_tanh_rn(const float* inn, const float* r, const float* hn,
                           float* o, int n) {
    int i = blockIdx.x * blockDim.x + threadIdx.x;
    if (i < n) o[i] = tanhf(inn[i] + r[i] * hn[i]);
}
// hnew = (1-z)*ng + z*h ; z = sigmoid(iz+hz)
__global__ void ew_zcomb(const float* iz, const float* hz, const float* ng,
                         const float* h, float* o, int n) {
    int i = blockIdx.x * blockDim.x + threadIdx.x;
    if (i < n) {
        float z = sigm(iz[i] + hz[i]);
        o[i] = (1.0f - z) * ng[i] + z * h[i];
    }
}
// h = s*h + (1-s)*g0 ; s = sigmoid(g)   (in-place on h)
__global__ void ew_gatemix(const float* g, const float* g0, float* h, int n) {
    int i = blockIdx.x * blockDim.x + threadIdx.x;
    if (i < n) {
        float s = sigm(g[i]);
        h[i] = s * h[i] + (1.0f - s) * g0[i];
    }
}
// out = s*his + (1-s)*h ; s = sigmoid(g)
__global__ void ew_timegate(const float* g, const float* his, const float* h,
                            float* o, int n) {
    int i = blockIdx.x * blockDim.x + threadIdx.x;
    if (i < n) {
        float s = sigm(g[i]);
        o[i] = s * his[i] + (1.0f - s) * h[i];
    }
}

// ---------------------------------------------------------------------------
extern "C" void kernel_launch(void* const* d_in, const int* in_sizes, int n_in,
                              void* d_out, int out_size, void* d_ws, size_t ws_size,
                              hipStream_t stream) {
    (void)in_sizes; (void)n_in; (void)out_size; (void)ws_size;

    const float* dyn  = (const float*)d_in[0];
    const float* erel = (const float*)d_in[1];
    const float* Wn1  = (const float*)d_in[2];
    const float* Wn2  = (const float*)d_in[3];
    const float* Wn3  = (const float*)d_in[4];
    const float* wih  = (const float*)d_in[5];   // [600,200]
    const float* whh  = (const float*)d_in[6];   // [600,200]
    const float* bih  = (const float*)d_in[7];   // [600]
    const float* bhh  = (const float*)d_in[8];   // [600]
    const float* ggw  = (const float*)d_in[9];   // [200,200]
    const float* ggb  = (const float*)d_in[10];  // [200]
    const float* tgw  = (const float*)d_in[11];
    const float* tgb  = (const float*)d_in[12];
    const int* srcA   = (const int*)d_in[13];    // [T,NE]
    const int* dstA   = (const int*)d_in[14];
    const int* etA    = (const int*)d_in[15];
    float* out        = (float*)d_out;           // [2, E, D]

    const size_t ED = (size_t)E_N * D_N;
    float* w    = (float*)d_ws;
    float* hn0  = w; w += ED;                 // l2norm(dynamic_emb) == gruh == init_h
    float* h    = w; w += ED;                 // evolving h
    float* agg  = w; w += ED;                 // edge aggregate
    float* xbuf = w; w += ED;                 // rgcn layer outputs / current_h
    float* t1   = w; w += ED;
    float* t2   = w; w += ED;
    float* t3   = w; w += ED;
    float* t4   = w; w += ED;
    float* r0   = w; w += (size_t)NR2_N * D_N;
    float* deg  = w; w += E_N;

    const dim3 gemm_grid((D_N + 63) / 64, (E_N + 63) / 64);
    const dim3 gemm_blk(128);
    const int ewg = (int)((ED + 255) / 256);
    const int edge_grid = NE_N / 8;             // 8 waves per 256-thread block

    #define GEMM(A_, B_, C_, TRANS_, MODE_, DEG_, BIAS_) \
        gemm_bf16_wmma<<<gemm_grid, gemm_blk, 0, stream>>>( \
            A_, B_, C_, E_N, D_N, D_N, D_N, TRANS_, MODE_, DEG_, BIAS_, RRELU_SLOPE)

    // h = init_h = gruh = l2norm(dynamic_emb); r0 = l2norm(emb_rel)
    l2norm_rows<<<(E_N + 7) / 8, 256, 0, stream>>>(dyn, hn0, E_N);
    hipMemcpyAsync(h, hn0, ED * sizeof(float), hipMemcpyDeviceToDevice, stream);
    l2norm_rows<<<(NR2_N + 7) / 8, 256, 0, stream>>>(erel, r0, NR2_N);

    for (int i = 0; i < T_N; ++i) {
        const int* s = srcA + (size_t)i * NE_N;
        const int* d = dstA + (size_t)i * NE_N;
        const int* t = etA  + (size_t)i * NE_N;

        // ---- RGCN layer 1: x1 = rrelu(norm * (segsum(gruh[s]+r0[t]) @ Wn1))
        hipMemsetAsync(agg, 0, ED * sizeof(float), stream);
        hipMemsetAsync(deg, 0, E_N * sizeof(float), stream);
        edge_agg<<<edge_grid, 256, 0, stream>>>(hn0, r0, s, d, t, agg, deg, NE_N, 1);
        GEMM(agg, Wn1, xbuf, 0, 1, deg, (const float*)nullptr);

        // ---- RGCN layer 2: x2 = rrelu(norm * (segsum(x1[s]+r0[t]) @ Wn2))
        hipMemsetAsync(agg, 0, ED * sizeof(float), stream);
        edge_agg<<<edge_grid, 256, 0, stream>>>(xbuf, r0, s, d, t, agg, deg, NE_N, 0);
        GEMM(agg, Wn2, xbuf, 0, 1, deg, (const float*)nullptr);

        // ---- current_h = l2norm(x2)  (in place)
        l2norm_rows<<<(E_N + 7) / 8, 256, 0, stream>>>(xbuf, xbuf, E_N);

        // ---- GRU cell (3 slices of width D against wih/whh rows)
        GEMM(xbuf, wih + 0 * D_N * D_N, t1, 1, 0, (const float*)nullptr, bih + 0);
        GEMM(h,    whh + 0 * D_N * D_N, t2, 1, 0, (const float*)nullptr, bhh + 0);
        ew_sigadd<<<ewg, 256, 0, stream>>>(t1, t2, t3, (int)ED);            // r
        GEMM(xbuf, wih + 2 * D_N * D_N, t1, 1, 0, (const float*)nullptr, bih + 2 * D_N);
        GEMM(h,    whh + 2 * D_N * D_N, t2, 1, 0, (const float*)nullptr, bhh + 2 * D_N);
        ew_tanh_rn<<<ewg, 256, 0, stream>>>(t1, t3, t2, t4, (int)ED);       // n
        GEMM(xbuf, wih + 1 * D_N * D_N, t1, 1, 0, (const float*)nullptr, bih + 1 * D_N);
        GEMM(h,    whh + 1 * D_N * D_N, t2, 1, 0, (const float*)nullptr, bhh + 1 * D_N);
        ew_zcomb<<<ewg, 256, 0, stream>>>(t1, t2, t4, h, t3, (int)ED);      // hnew
        l2norm_rows<<<(E_N + 7) / 8, 256, 0, stream>>>(t3, h, E_N);         // h = l2norm

        // ---- gruh gate: h = s*h + (1-s)*gruh, s = sigmoid(h @ ggw + ggb)
        GEMM(h, ggw, t1, 0, 0, (const float*)nullptr, ggb);
        ew_gatemix<<<ewg, 256, 0, stream>>>(t1, hn0, h, (int)ED);

        if (i == T_N - 1) {
            // ---- his = l2norm(rrelu(norm * (segsum(init_h[s]+r0[t]) @ Wn3)))
            hipMemsetAsync(agg, 0, ED * sizeof(float), stream);
            edge_agg<<<edge_grid, 256, 0, stream>>>(hn0, r0, s, d, t, agg, deg, NE_N, 0);
            GEMM(agg, Wn3, xbuf, 0, 1, deg, (const float*)nullptr);
            l2norm_rows<<<(E_N + 7) / 8, 256, 0, stream>>>(xbuf, t1, E_N);  // his
            // ---- time gate: out1 = s*his + (1-s)*h, s = sigmoid(his @ tgw + tgb)
            GEMM(t1, tgw, t2, 0, 0, (const float*)nullptr, tgb);
            ew_timegate<<<ewg, 256, 0, stream>>>(t2, t1, h, out + ED, (int)ED);
            hipMemcpyAsync(out, h, ED * sizeof(float), hipMemcpyDeviceToDevice, stream);
        }
    }
    #undef GEMM
}
